// PTRnet_38070590112047
// MI455X (gfx1250) — matrix operand order repaired and checked by reading, not compiled, not run on hardware
//
#include <hip/hip_runtime.h>
#include <hip/hip_bf16.h>
#include <math.h>

// ---------------------------------------------------------------------------
// Model dims
// ---------------------------------------------------------------------------
#define Bb   4
#define Ll   2048
#define Mrows (Bb * Ll)      // 8192
#define DMm  512
#define DIi  1024
#define DSs  16
#define DCc  4
#define DTRr 32
#define NLl  4
#define NOUT 19

typedef __attribute__((ext_vector_type(16))) __bf16 v16bf;
typedef __attribute__((ext_vector_type(8)))  float  v8f;
typedef __attribute__((ext_vector_type(2)))  __bf16 bf2;

// ---------------------------------------------------------------------------
// Embedding: x[b,l,d] = (se + te) * (se != 0),  se = sum of 3 token embeddings
// ---------------------------------------------------------------------------
__global__ void embed_kernel(const int* __restrict__ rna,
                             const int* __restrict__ tissue_id,
                             const float* __restrict__ seq_emb,
                             const float* __restrict__ tissue_emb,
                             float* __restrict__ x)
{
    size_t i = (size_t)blockIdx.x * blockDim.x + threadIdx.x;
    if (i >= (size_t)Mrows * DMm) return;
    int d   = (int)(i & (DMm - 1));
    int row = (int)(i >> 9);           // DMm = 512 = 2^9
    int b   = row >> 11;               // Ll = 2048 = 2^11
    const int* t = rna + (size_t)row * 3;
    float se = seq_emb[(size_t)t[0] * DMm + d]
             + seq_emb[(size_t)t[1] * DMm + d]
             + seq_emb[(size_t)t[2] * DMm + d];
    float v = 0.0f;
    if (se != 0.0f)
        v = se + tissue_emb[(size_t)tissue_id[b] * DMm + d];
    x[i] = v;
}

// ---------------------------------------------------------------------------
// Generic f32 -> bf16 cast
// ---------------------------------------------------------------------------
__global__ void cast_bf16_kernel(const float* __restrict__ src,
                                 __bf16* __restrict__ dst, size_t n)
{
    size_t i = (size_t)blockIdx.x * blockDim.x + threadIdx.x;
    if (i < n) dst[i] = (__bf16)src[i];
}

// ---------------------------------------------------------------------------
// WMMA bf16 GEMM:  C[M,N] (f32, row-major) = A[M,K] (bf16 row-major)
//                                          x W[N,K]^T (bf16 row-major)
// Block = 128 threads (4 waves). Block tile = 64 (M) x 64 (N), K chunk = 64.
// Wave w owns rows [w*16, w*16+16) x all four 16-wide N tiles:
//   8 WMMAs per wave per barrier round (2 k-halves x 4 n-tiles).
// A/B tiles staged into LDS via gfx1250 GLOBAL_LOAD_ASYNC_TO_LDS_B128
// (ASYNCcnt), completed with s_wait_asynccnt + workgroup barrier.
// M must be a multiple of 64, K a multiple of 32; N may be ragged (guarded).
// ---------------------------------------------------------------------------
__global__ void gemm_bf16_wmma(const __bf16* __restrict__ A,
                               const __bf16* __restrict__ W,
                               float* __restrict__ C,
                               int M, int N, int K)
{
    const int tid  = threadIdx.x;
    const int wave = tid >> 5;
    const int lane = tid & 31;
    const int m0   = blockIdx.x * 64;
    const int n0b  = blockIdx.y * 64;
    const int h    = lane >> 4;      // half-wave
    const int mloc = lane & 15;
    const int nloc = lane & 15;

    __shared__ __align__(16) __bf16 As[64 * 64];   // [row][k] , k-chunk of 64
    __shared__ __align__(16) __bf16 Bs[64 * 64];   // [nrow][k]

    // Pre-zero B tile once if this block's N range is ragged; async loads are
    // EXEC-masked for invalid rows so those rows stay zero for every k chunk.
    if (n0b + 64 > N) {
#pragma unroll
        for (int c = 0; c < 4; ++c)
            *reinterpret_cast<uint4*>(&Bs[tid * 32 + c * 8]) =
                uint4{0u, 0u, 0u, 0u};
    }
    __syncthreads();

    v8f acc0 = {}, acc1 = {}, acc2 = {}, acc3 = {};

    // per-thread staging geometry: thread -> (row, 32-element column half)
    const int lr = tid >> 1;           // 0..63
    const int lc = (tid & 1) * 32;     // 0 or 32

    for (int k0 = 0; k0 < K; k0 += 64) {
        // ---- stage A (64 x 64 bf16) via async copies: 4 x B128 per thread
#pragma unroll
        for (int c = 0; c < 32; c += 8) {
            int col = lc + c;
            if (k0 + col < K) {
                unsigned long long ga =
                    (unsigned long long)(A + (size_t)(m0 + lr) * K + k0 + col);
                unsigned lo = (unsigned)(unsigned long long)(void*)&As[lr * 64 + col];
                asm volatile("global_load_async_to_lds_b128 %0, %1, off"
                             :: "v"(lo), "v"(ga) : "memory");
            }
        }
        // ---- stage B (rows n0b..n0b+63 of W) via async copies
        {
            int nrow = n0b + lr;
            if (nrow < N) {
#pragma unroll
                for (int c = 0; c < 32; c += 8) {
                    int col = lc + c;
                    if (k0 + col < K) {
                        unsigned long long gb =
                            (unsigned long long)(W + (size_t)nrow * K + k0 + col);
                        unsigned lo =
                            (unsigned)(unsigned long long)(void*)&Bs[lr * 64 + col];
                        asm volatile("global_load_async_to_lds_b128 %0, %1, off"
                                     :: "v"(lo), "v"(gb) : "memory");
                    }
                }
            }
        }
        asm volatile("s_wait_asynccnt 0x0" ::: "memory");
        __syncthreads();

        // ---- compute: 2 k-halves x 4 n-tiles of WMMA
#pragma unroll
        for (int kk = 0; kk < 64; kk += 32) {
            if (k0 + kk < K) {
                v16bf a;
#pragma unroll
                for (int j = 0; j < 8; ++j) {
                    int ka = ((j >> 2) << 4) + (h << 3) + ((j & 3) << 1);
                    bf2 pa = *reinterpret_cast<const bf2*>(
                        &As[(wave * 16 + mloc) * 64 + kk + ka]);
                    a[2 * j]     = pa.x;
                    a[2 * j + 1] = pa.y;
                }
#pragma unroll
                for (int t = 0; t < 4; ++t) {
                    v16bf b;
#pragma unroll
                    for (int j = 0; j < 8; ++j) {
                        int kb = ((j >> 2) << 4) + (h << 3) + ((j & 3) << 1);
                        bf2 pb = *reinterpret_cast<const bf2*>(
                            &Bs[(t * 16 + nloc) * 64 + kk + kb]);
                        b[2 * j]     = pb.x;
                        b[2 * j + 1] = pb.y;
                    }
                    v8f* accp = (t == 0) ? &acc0 : (t == 1) ? &acc1
                              : (t == 2) ? &acc2 : &acc3;
                    *accp = __builtin_amdgcn_wmma_f32_16x16x32_bf16(
                        false, a, false, b, (short)0, *accp, false, false);
                }
            }
        }
        __syncthreads();
    }

    // ---- store: VGPR r of each acc holds (M = r + 8*h, N = lane&15)
#pragma unroll
    for (int t = 0; t < 4; ++t) {
        int n = n0b + t * 16 + nloc;
        if (n < N) {
            const v8f& acc = (t == 0) ? acc0 : (t == 1) ? acc1
                           : (t == 2) ? acc2 : acc3;
#pragma unroll
            for (int r = 0; r < 8; ++r) {
                int m = m0 + wave * 16 + r + (h << 3);
                C[(size_t)m * N + n] = acc[r];
            }
        }
    }
}

// ---------------------------------------------------------------------------
// Causal depthwise conv (DC=4) + bias + SiLU over xin = xz[:, :DI]
// Writes f32 (for scan) and bf16 (for x_proj GEMM).
// ---------------------------------------------------------------------------
__global__ void conv_silu_kernel(const float* __restrict__ xz,
                                 const float* __restrict__ cw,  // [DI,4]
                                 const float* __restrict__ cb,  // [DI]
                                 float* __restrict__ xin_f,
                                 __bf16* __restrict__ xin_bf)
{
    size_t i = (size_t)blockIdx.x * blockDim.x + threadIdx.x;
    if (i >= (size_t)Mrows * DIi) return;
    int di  = (int)(i & (DIi - 1));
    int row = (int)(i >> 10);          // DIi = 1024 = 2^10
    int b   = row >> 11;
    int l   = row & (Ll - 1);
    float acc = cb[di];
#pragma unroll
    for (int j = 0; j < DCc; ++j) {
        int ll = l - (DCc - 1) + j;
        if (ll >= 0)
            acc += xz[((size_t)(b * Ll + ll)) * (2 * DIi) + di] * cw[di * DCc + j];
    }
    float s = acc / (1.0f + expf(-acc));
    xin_f[i]  = s;
    xin_bf[i] = (__bf16)s;
}

// ---------------------------------------------------------------------------
// Extract dt (first DTR cols of xdbl) and cast to bf16 for the dt_proj GEMM
// ---------------------------------------------------------------------------
__global__ void dt_cast_kernel(const float* __restrict__ xdbl,
                               __bf16* __restrict__ dt_bf)
{
    size_t i = (size_t)blockIdx.x * blockDim.x + threadIdx.x;
    if (i >= (size_t)Mrows * DTRr) return;
    int j   = (int)(i & (DTRr - 1));
    int row = (int)(i >> 5);
    dt_bf[i] = (__bf16)xdbl[(size_t)row * (DTRr + 2 * DSs) + j];
}

// ---------------------------------------------------------------------------
// Selective scan: per (b, channel) sequential recurrence over L.
// delta = softplus(delta_pre + dtb); A = -exp(Alog);
// h = h*exp(delta*A) + delta*x*B ; y = <h,C> + x*D ; y *= silu(z)
// Grid: (DI/256, B), block 256. B/C broadcast via LDS.
// ---------------------------------------------------------------------------
__global__ void scan_kernel(const float* __restrict__ delta_pre, // [M,DI]
                            const float* __restrict__ xin_f,     // [M,DI]
                            const float* __restrict__ xz,        // [M,2DI]
                            const float* __restrict__ xdbl,      // [M,64]
                            const float* __restrict__ Alog,      // [DI,DS]
                            const float* __restrict__ dtb,       // [DI]
                            const float* __restrict__ Dp,        // [DI]
                            __bf16* __restrict__ ybf)            // [M,DI]
{
    const int b  = blockIdx.y;
    const int di = blockIdx.x * blockDim.x + threadIdx.x;

    float Avec[DSs];
#pragma unroll
    for (int s = 0; s < DSs; ++s)
        Avec[s] = -expf(Alog[(size_t)di * DSs + s]);
    const float dtbv = dtb[di];
    const float Dv   = Dp[di];

    float hst[DSs];
#pragma unroll
    for (int s = 0; s < DSs; ++s) hst[s] = 0.0f;

    __shared__ float Bsh[DSs];
    __shared__ float Csh[DSs];

    for (int l = 0; l < Ll; ++l) {
        size_t row = (size_t)b * Ll + l;
        if (threadIdx.x < 2 * DSs) {
            float v = xdbl[row * (DTRr + 2 * DSs) + DTRr + threadIdx.x];
            if (threadIdx.x < DSs) Bsh[threadIdx.x] = v;
            else                   Csh[threadIdx.x - DSs] = v;
        }
        __syncthreads();

        float t  = delta_pre[row * DIi + di] + dtbv;
        float dl = (t > 20.0f) ? t : log1pf(expf(t));
        float xv = xin_f[row * DIi + di];
        float dx = dl * xv;
        float y  = 0.0f;
#pragma unroll
        for (int s = 0; s < DSs; ++s) {
            float dA = expf(dl * Avec[s]);
            hst[s] = hst[s] * dA + dx * Bsh[s];
            y += hst[s] * Csh[s];
        }
        y += xv * Dv;
        float zv = xz[row * (2 * DIi) + DIi + di];
        y *= zv / (1.0f + expf(-zv));
        ybf[row * DIi + di] = (__bf16)y;

        __syncthreads();
    }
}

// ---------------------------------------------------------------------------
// Head: bias + grouped softmax over {4,5,3,7}
// ---------------------------------------------------------------------------
__global__ void head_softmax_kernel(const float* __restrict__ logits, // [M,19]
                                    const float* __restrict__ hb,     // [19]
                                    float* __restrict__ out)          // [M,19]
{
    int i = blockIdx.x * blockDim.x + threadIdx.x;
    if (i >= Mrows) return;
    float v[NOUT];
#pragma unroll
    for (int o = 0; o < NOUT; ++o)
        v[o] = logits[(size_t)i * NOUT + o] + hb[o];
    const int sizes[4] = {4, 5, 3, 7};
    int off = 0;
    for (int g = 0; g < 4; ++g) {
        int sz = sizes[g];
        float mx = v[off];
        for (int k = 1; k < sz; ++k) mx = fmaxf(mx, v[off + k]);
        float sum = 0.0f;
        for (int k = 0; k < sz; ++k) {
            float e = expf(v[off + k] - mx);
            v[off + k] = e;
            sum += e;
        }
        float inv = 1.0f / sum;
        for (int k = 0; k < sz; ++k)
            out[(size_t)i * NOUT + off + k] = v[off + k] * inv;
        off += sz;
    }
}

// ---------------------------------------------------------------------------
// Host launcher
// ---------------------------------------------------------------------------
extern "C" void kernel_launch(void* const* d_in, const int* in_sizes, int n_in,
                              void* d_out, int out_size, void* d_ws, size_t ws_size,
                              hipStream_t stream)
{
    const int*   rna        = (const int*)  d_in[0];
    const int*   tissue_id  = (const int*)  d_in[1];
    /* d_in[2] = seq_lengths: unused by the reference */
    const float* seq_emb    = (const float*)d_in[3];
    const float* tissue_emb = (const float*)d_in[4];
    const float* in_proj_w  = (const float*)d_in[5];
    const float* conv_w     = (const float*)d_in[6];
    const float* conv_b     = (const float*)d_in[7];
    const float* x_proj_w   = (const float*)d_in[8];
    const float* dt_proj_w  = (const float*)d_in[9];
    const float* dt_proj_b  = (const float*)d_in[10];
    const float* A_log      = (const float*)d_in[11];
    const float* Dp         = (const float*)d_in[12];
    const float* out_proj_w = (const float*)d_in[13];
    const float* head_w     = (const float*)d_in[14];
    const float* head_b     = (const float*)d_in[15];
    float* out = (float*)d_out;

    // ---- carve workspace ----
    char*  ws  = (char*)d_ws;
    size_t off = 0;
    auto take = [&](size_t bytes) -> char* {
        char* p = ws + off;
        off += (bytes + 255) & ~(size_t)255;
        return p;
    };
    const size_t M = Mrows;
    float*  X      = (float*) take(M * DMm * sizeof(float));
    __bf16* XBF    = (__bf16*)take(M * DMm * sizeof(__bf16));
    float*  XZ     = (float*) take(M * 2 * DIi * sizeof(float));
    float*  XIN    = (float*) take(M * DIi * sizeof(float));
    __bf16* XINBF  = (__bf16*)take(M * DIi * sizeof(__bf16));
    float*  XDBL   = (float*) take(M * (DTRr + 2 * DSs) * sizeof(float));
    __bf16* DTBF   = (__bf16*)take(M * DTRr * sizeof(__bf16));
    float*  DELTA  = (float*) take(M * DIi * sizeof(float));
    __bf16* YBF    = (__bf16*)take(M * DIi * sizeof(__bf16));
    float*  LOGITS = (float*) take(M * NOUT * sizeof(float));
    __bf16* W_INP  = (__bf16*)take((size_t)NLl * 2 * DIi * DMm * sizeof(__bf16));
    __bf16* W_XP   = (__bf16*)take((size_t)NLl * (DTRr + 2 * DSs) * DIi * sizeof(__bf16));
    __bf16* W_DT   = (__bf16*)take((size_t)NLl * DIi * DTRr * sizeof(__bf16));
    __bf16* W_OP   = (__bf16*)take((size_t)NLl * DMm * DIi * sizeof(__bf16));
    __bf16* W_HD   = (__bf16*)take((size_t)NOUT * DMm * sizeof(__bf16));

    auto cast = [&](const float* src, __bf16* dst, size_t n) {
        cast_bf16_kernel<<<dim3((unsigned)((n + 255) / 256)), dim3(256), 0, stream>>>(src, dst, n);
    };
    auto gemm = [&](const __bf16* A, const __bf16* W, float* C, int n, int k) {
        gemm_bf16_wmma<<<dim3((unsigned)(M / 64), (unsigned)((n + 63) / 64)),
                         dim3(128), 0, stream>>>(A, W, C, (int)M, n, k);
    };

    // ---- one-time weight casts to bf16 ----
    cast(in_proj_w,  W_INP, (size_t)NLl * 2 * DIi * DMm);
    cast(x_proj_w,   W_XP,  (size_t)NLl * (DTRr + 2 * DSs) * DIi);
    cast(dt_proj_w,  W_DT,  (size_t)NLl * DIi * DTRr);
    cast(out_proj_w, W_OP,  (size_t)NLl * DMm * DIi);
    cast(head_w,     W_HD,  (size_t)NOUT * DMm);

    // ---- embedding ----
    embed_kernel<<<dim3((unsigned)((M * DMm + 255) / 256)), dim3(256), 0, stream>>>(
        rna, tissue_id, seq_emb, tissue_emb, X);

    // ---- Mamba layers ----
    for (int l = 0; l < NLl; ++l) {
        const __bf16* wi  = W_INP + (size_t)l * 2 * DIi * DMm;
        const float*  cw  = conv_w  + (size_t)l * DIi * DCc;
        const float*  cb  = conv_b  + (size_t)l * DIi;
        const __bf16* wxp = W_XP  + (size_t)l * (DTRr + 2 * DSs) * DIi;
        const __bf16* wdt = W_DT  + (size_t)l * DIi * DTRr;
        const float*  dtb = dt_proj_b + (size_t)l * DIi;
        const float*  al  = A_log + (size_t)l * DIi * DSs;
        const float*  dv  = Dp    + (size_t)l * DIi;
        const __bf16* wop = W_OP  + (size_t)l * DMm * DIi;

        // in_proj: [M,512] x [2048,512]^T -> [M,2048]
        cast(X, XBF, M * DMm);
        gemm(XBF, wi, XZ, 2 * DIi, DMm);

        // conv + SiLU
        conv_silu_kernel<<<dim3((unsigned)((M * DIi + 255) / 256)), dim3(256), 0, stream>>>(
            XZ, cw, cb, XIN, XINBF);

        // x_proj: [M,1024] x [64,1024]^T -> [M,64]
        gemm(XINBF, wxp, XDBL, DTRr + 2 * DSs, DIi);

        // dt cast + dt_proj: [M,32] x [1024,32]^T -> [M,1024]
        dt_cast_kernel<<<dim3((unsigned)((M * DTRr + 255) / 256)), dim3(256), 0, stream>>>(
            XDBL, DTBF);
        gemm(DTBF, wdt, DELTA, DIi, DTRr);

        // selective scan (sequential in L)
        scan_kernel<<<dim3(DIi / 256, Bb), dim3(256), 0, stream>>>(
            DELTA, XIN, XZ, XDBL, al, dtb, dv, YBF);

        // out_proj: [M,1024] x [512,1024]^T -> [M,512] (becomes x)
        gemm(YBF, wop, X, DMm, DIi);
    }

    // ---- head + grouped softmax ----
    cast(X, XBF, M * DMm);
    gemm(XBF, W_HD, LOGITS, NOUT, DMm);
    head_softmax_kernel<<<dim3((unsigned)((M + 255) / 256)), dim3(256), 0, stream>>>(
        LOGITS, head_b, out);
}
